// GPT2Attention_26912265077078
// MI455X (gfx1250) — matrix operand order repaired and checked by reading, not compiled, not run on hardware
//
#include <hip/hip_runtime.h>
#include <hip/hip_bf16.h>

// ---------------------------------------------------------------------------
// GPT-2 attention block for MI455X (gfx1250, wave32, WMMA bf16 + TDM path)
// ---------------------------------------------------------------------------

typedef __attribute__((ext_vector_type(16))) __bf16 v16bf_t;
typedef __attribute__((ext_vector_type(8)))  float  v8f_t;
typedef __attribute__((ext_vector_type(8)))  unsigned short us8_t;
typedef __attribute__((ext_vector_type(4)))  unsigned int u32x4_t;
typedef __attribute__((ext_vector_type(8)))  int i32x8_t;
typedef __attribute__((ext_vector_type(4)))  int i32x4_t;

#define E_DIM  1024
#define S_LEN  2048
#define B_SZ   2
#define H_CNT  16
#define D_DIM  64
#define M_ROWS (B_SZ * S_LEN)   // 4096

__device__ __forceinline__ unsigned short f2bf(float x) {
  unsigned int u = __builtin_bit_cast(unsigned int, x);
  u += 0x7fffu + ((u >> 16) & 1u);          // round-to-nearest-even
  return (unsigned short)(u >> 16);
}

__device__ __forceinline__ v8f_t wmma_bf16(v16bf_t a, v16bf_t b, v8f_t c) {
  // D = A(16x32 bf16) * B(32x16 bf16) + C(16x16 f32)
  return __builtin_amdgcn_wmma_f32_16x16x32_bf16(
      /*neg_a=*/false, a, /*neg_b=*/false, b,
      /*c_mod=*/(short)0, c, /*reuse_a=*/false, /*reuse_b=*/false);
}

union FragU { us8_t u[2]; v16bf_t v; };

// A-fragment (16x32, bf16): lane L holds row (L&15); halves 0..7 -> K = kb..kb+7,
// halves 8..15 -> K = 16+kb..16+kb+7, kb = (L>>4)*8.  base points at (row0, k0).
__device__ __forceinline__ v16bf_t load_frag_a(const unsigned short* base, int stride, int lane) {
  const int row = lane & 15;
  const int kb  = (lane >> 4) * 8;
  FragU f;
  f.u[0] = *(const us8_t*)(base + row * stride + kb);
  f.u[1] = *(const us8_t*)(base + row * stride + kb + 16);
  return f.v;
}

// B-fragment (32x16, bf16) from a B^T [n][k] tile: lane L holds column (L&15);
// halves 0..7 -> K = kb..kb+7, halves 8..15 -> K = 16+kb..16+kb+7, kb = (L>>4)*8.  base points at (n0, k0) of B^T.
__device__ __forceinline__ v16bf_t load_frag_b(const unsigned short* base, int stride, int lane) {
  const int n  = lane & 15;
  const int kb = (lane >> 4) * 8;
  FragU f;
  f.u[0] = *(const us8_t*)(base + n * stride + kb);
  f.u[1] = *(const us8_t*)(base + n * stride + kb + 16);
  return f.v;
}

// ---------------------------------------------------------------------------
// Tensor Data Mover: DMA one 2D tile (bf16 elements) global -> LDS.
// D# packing per CDNA5 ISA (group0 + group1, 2D, no pad,
// no gather, no iterate, no multicast).  tensor dims = tile dims (tiles are
// in-bounds by construction).  Units: dims/strides in elements, addrs bytes.
// This toolchain exposes the 6-arg builtin (groups 2/3 + extra group, zeroed).
// ---------------------------------------------------------------------------
__device__ __forceinline__ void tdm_load_2d(unsigned lds_off, const void* gaddr,
                                            unsigned tile_d0, unsigned tile_d1,
                                            unsigned stride0) {
  const unsigned long long ga = (unsigned long long)(size_t)gaddr;
  u32x4_t g0;
  g0[0] = 1u;                                         // count=1, user mode
  g0[1] = lds_off;                                    // lds_addr (bytes)
  g0[2] = (unsigned)(ga & 0xffffffffu);               // global_addr[31:0]
  g0[3] = (unsigned)((ga >> 32) & 0x01ffffffu)        // global_addr[56:32]
        | (2u << 30);                                 // type = 2 ("image")
  i32x8_t g1;
  g1[0] = (int)(1u << 16);                            // data_size = 1 -> 2 bytes
  g1[1] = (int)((tile_d0 & 0xffffu) << 16);           // tensor_dim0[15:0]
  g1[2] = (int)(((tile_d0 >> 16) & 0xffffu) | ((tile_d1 & 0xffffu) << 16));
  g1[3] = (int)(((tile_d1 >> 16) & 0xffffu) | ((tile_d0 & 0xffffu) << 16)); // tile_dim0
  g1[4] = (int)(tile_d1 & 0xffffu);                   // tile_dim1 (tile_dim2 = 0)
  g1[5] = (int)stride0;                               // tensor_dim0_stride[31:0]
  g1[6] = 0;                                          // stride0[47:32], stride1[15:0]
  g1[7] = 0;                                          // stride1[47:16]
  const i32x4_t gz4 = {0, 0, 0, 0};
  const i32x8_t gz8 = {0, 0, 0, 0, 0, 0, 0, 0};
  __builtin_amdgcn_tensor_load_to_lds(g0, g1, gz4, gz4, gz8, 0);
}

__device__ __forceinline__ unsigned lds_off_of(const void* p) {
  return (unsigned)(size_t)p;   // LDS aperture: low 32 bits = offset in wave LDS
}

// ---------------------------------------------------------------------------
// Precision-conversion kernels
// ---------------------------------------------------------------------------
__global__ void cvt_f32_bf16_kernel(const float* __restrict__ src,
                                    unsigned short* __restrict__ dst, int n) {
  for (int i = blockIdx.x * blockDim.x + threadIdx.x; i < n; i += gridDim.x * blockDim.x)
    dst[i] = f2bf(src[i]);
}

// src [K][N] f32  ->  dst [N][K] bf16   (weights stored transposed for B^T tiles)
__global__ void transpose_cvt_kernel(const float* __restrict__ src,
                                     unsigned short* __restrict__ dst, int K, int N) {
  const int total = K * N;
  for (int i = blockIdx.x * blockDim.x + threadIdx.x; i < total; i += gridDim.x * blockDim.x) {
    const int k = i / N, n = i - k * N;
    dst[n * K + k] = f2bf(src[i]);
  }
}

// ---------------------------------------------------------------------------
// Tiled GEMM: C[M][N] = A[M][1024](bf16) * Bt[N][1024]^T(bf16) + bias
// Block tile 128x128, 8 waves (4x2), wave tile 32x64 -> 2x4 WMMA accumulators.
// TDM double-buffered K staging (wave 0 drives the DMA pipeline).
// MODE 0: QKV epilogue (scatter bf16 into Q[b,h,s,d], K[b,h,s,d], Vt[b,h,d,s])
// MODE 1: f32 output + bias to d_out
// ---------------------------------------------------------------------------
template <int MODE>
__global__ __launch_bounds__(256) void gemm128_kernel(
    const unsigned short* __restrict__ A,
    const unsigned short* __restrict__ Bt,
    const float* __restrict__ bias,
    unsigned short* __restrict__ qws,
    unsigned short* __restrict__ kws,
    unsigned short* __restrict__ vtws,
    float* __restrict__ out)
{
  __shared__ __align__(16) unsigned short ldsA[2][128 * 64];
  __shared__ __align__(16) unsigned short ldsB[2][128 * 64];

  const int tid = threadIdx.x, wid = tid >> 5, lane = tid & 31;
  const int lane15 = lane & 15, hi8 = (lane >> 4) * 8;
  const int m0 = blockIdx.y * 128, n0 = blockIdx.x * 128;
  const int wm = (wid >> 1) * 32, wn = (wid & 1) * 64;

  const unsigned short* Ab = A  + m0 * 1024;
  const unsigned short* Bb = Bt + n0 * 1024;

  v8f_t acc[2][4] = {};

  constexpr int NT = 1024 / 64;   // 16 K-tiles

  if (wid == 0) {   // prologue: DMA tile 0 into buffer 0
    tdm_load_2d(lds_off_of(&ldsA[0][0]), Ab, 64, 128, 1024);
    tdm_load_2d(lds_off_of(&ldsB[0][0]), Bb, 64, 128, 1024);
  }

  for (int i = 0; i < NT; ++i) {
    const int cur = i & 1;
    if (wid == 0) {
      if (i + 1 < NT) {   // prefetch tile i+1 into the other buffer
        const int k0 = (i + 1) * 64;
        tdm_load_2d(lds_off_of(&ldsA[cur ^ 1][0]), Ab + k0, 64, 128, 1024);
        tdm_load_2d(lds_off_of(&ldsB[cur ^ 1][0]), Bb + k0, 64, 128, 1024);
        __builtin_amdgcn_s_wait_tensorcnt(2);   // tile i done; i+1 in flight
      } else {
        __builtin_amdgcn_s_wait_tensorcnt(0);
      }
    }
    __syncthreads();   // tile i visible to all waves

    const unsigned short* tA = &ldsA[cur][0];
    const unsigned short* tB = &ldsB[cur][0];
#pragma unroll
    for (int ks = 0; ks < 64; ks += 32) {
      v16bf_t a0 = load_frag_a(tA + (wm +  0) * 64 + ks, 64, lane);
      v16bf_t a1 = load_frag_a(tA + (wm + 16) * 64 + ks, 64, lane);
      v16bf_t b0 = load_frag_b(tB + (wn +  0) * 64 + ks, 64, lane);
      v16bf_t b1 = load_frag_b(tB + (wn + 16) * 64 + ks, 64, lane);
      v16bf_t b2 = load_frag_b(tB + (wn + 32) * 64 + ks, 64, lane);
      v16bf_t b3 = load_frag_b(tB + (wn + 48) * 64 + ks, 64, lane);
      acc[0][0] = wmma_bf16(a0, b0, acc[0][0]);
      acc[0][1] = wmma_bf16(a0, b1, acc[0][1]);
      acc[0][2] = wmma_bf16(a0, b2, acc[0][2]);
      acc[0][3] = wmma_bf16(a0, b3, acc[0][3]);
      acc[1][0] = wmma_bf16(a1, b0, acc[1][0]);
      acc[1][1] = wmma_bf16(a1, b1, acc[1][1]);
      acc[1][2] = wmma_bf16(a1, b2, acc[1][2]);
      acc[1][3] = wmma_bf16(a1, b3, acc[1][3]);
    }
    __syncthreads();   // done reading buf cur before DMA reuses it at i+2
  }

#pragma unroll
  for (int mt = 0; mt < 2; ++mt)
#pragma unroll
    for (int nt = 0; nt < 4; ++nt) {
      const int n  = n0 + wn + nt * 16 + lane15;
      const float bv = bias[n];
#pragma unroll
      for (int r = 0; r < 8; ++r) {
        const int   m = m0 + wm + mt * 16 + hi8 + r;
        const float v = acc[mt][nt][r] + bv;
        if (MODE == 0) {
          const int sec = n >> 10, e = n & 1023, h = e >> 6, d = e & 63;
          const int b = m >> 11, s = m & 2047;
          const int bh = b * H_CNT + h;
          const unsigned short hv = f2bf(v);
          if (sec == 0)      qws [(bh * S_LEN + s) * D_DIM + d] = hv;
          else if (sec == 1) kws [(bh * S_LEN + s) * D_DIM + d] = hv;
          else               vtws[(bh * D_DIM + d) * S_LEN + s] = hv;
        } else {
          out[m * 1024 + n] = v;
        }
      }
    }
}

// ---------------------------------------------------------------------------
// Flash-attention: one (b,h) + 128 query rows per block; 8 waves x 16 rows.
// TDM double-buffers 64-key K/V tiles; online softmax; O accumulated via WMMA.
// ---------------------------------------------------------------------------
__global__ __launch_bounds__(256) void attn_fa_kernel(
    const unsigned short* __restrict__ Q,    // [BH][S][D] bf16
    const unsigned short* __restrict__ Kw,   // [BH][S][D] bf16
    const unsigned short* __restrict__ Vt,   // [BH][D][S] bf16
    unsigned short* __restrict__ Ow)         // [B][S][E]  bf16 (heads re-merged)
{
  __shared__ __align__(16) unsigned short ldsK[2][64 * 64];    // [s_k][d]
  __shared__ __align__(16) unsigned short ldsV[2][64 * 64];    // [d][s_k]
  __shared__ __align__(16) unsigned short ldsP[8 * 16 * 64];   // per-wave P patch

  const int tid = threadIdx.x, wid = tid >> 5, lane = tid & 31;
  const int lane15 = lane & 15, hi8 = (lane >> 4) * 8;
  const int bh = blockIdx.y;                  // 0..31
  const int q0 = blockIdx.x * 128 + wid * 16; // this wave's 16 query rows

  const unsigned short* Qb = Q  + bh * S_LEN * D_DIM;
  const unsigned short* Kb = Kw + bh * S_LEN * D_DIM;
  const unsigned short* Vb = Vt + bh * D_DIM * S_LEN;

  // Q A-fragments (K-dim = d, 64 = 2 x 32) held in registers for all key blocks
  const v16bf_t qa0 = load_frag_a(Qb + q0 * D_DIM,      D_DIM, lane);
  const v16bf_t qa1 = load_frag_a(Qb + q0 * D_DIM + 32, D_DIM, lane);

  v8f_t o[4] = {};
  float mr[8], lr[8];
#pragma unroll
  for (int r = 0; r < 8; ++r) { mr[r] = -3.0e38f; lr[r] = 0.f; }

  unsigned short* Pw = ldsP + wid * (16 * 64);

  constexpr int NKB = S_LEN / 64;   // 32 key blocks

  if (wid == 0) {   // prologue DMA: key block 0
    tdm_load_2d(lds_off_of(&ldsK[0][0]), Kb, 64, 64, 64);     // contiguous rows
    tdm_load_2d(lds_off_of(&ldsV[0][0]), Vb, 64, 64, 2048);   // strided d-rows
  }

  for (int i = 0; i < NKB; ++i) {
    const int cur = i & 1;
    if (wid == 0) {
      if (i + 1 < NKB) {
        const int kb = (i + 1) * 64;
        tdm_load_2d(lds_off_of(&ldsK[cur ^ 1][0]), Kb + kb * D_DIM, 64, 64, 64);
        tdm_load_2d(lds_off_of(&ldsV[cur ^ 1][0]), Vb + kb,         64, 64, 2048);
        __builtin_amdgcn_s_wait_tensorcnt(2);
      } else {
        __builtin_amdgcn_s_wait_tensorcnt(0);
      }
    }
    __syncthreads();

    const unsigned short* tK = &ldsK[cur][0];
    const unsigned short* tV = &ldsV[cur][0];

    // scores S = Q * K^T  (16 q-rows x 64 keys, as 4 16x16 tiles)
    v8f_t sc[4];
#pragma unroll
    for (int nt = 0; nt < 4; ++nt) {
      v8f_t z = {};
      z = wmma_bf16(qa0, load_frag_b(tK + nt * 16 * 64,      64, lane), z);
      z = wmma_bf16(qa1, load_frag_b(tK + nt * 16 * 64 + 32, 64, lane), z);
      sc[nt] = z;
    }
#pragma unroll
    for (int nt = 0; nt < 4; ++nt)
#pragma unroll
      for (int r = 0; r < 8; ++r) sc[nt][r] *= 0.125f;   // 1/sqrt(64)

    // online softmax: a row's 16 columns live in one 16-lane group
    float alpha[8];
#pragma unroll
    for (int r = 0; r < 8; ++r) {
      float t = fmaxf(fmaxf(sc[0][r], sc[1][r]), fmaxf(sc[2][r], sc[3][r]));
#pragma unroll
      for (int off = 1; off < 16; off <<= 1) t = fmaxf(t, __shfl_xor(t, off, 16));
      const float nm = fmaxf(mr[r], t);
      alpha[r] = __expf(mr[r] - nm);
      mr[r] = nm;
      float rs = 0.f;
#pragma unroll
      for (int nt = 0; nt < 4; ++nt) {
        const float p = __expf(sc[nt][r] - nm);
        sc[nt][r] = p;
        rs += p;
      }
#pragma unroll
      for (int off = 1; off < 16; off <<= 1) rs += __shfl_xor(rs, off, 16);
      lr[r] = lr[r] * alpha[r] + rs;
    }
#pragma unroll
    for (int d4 = 0; d4 < 4; ++d4)
#pragma unroll
      for (int r = 0; r < 8; ++r) o[d4][r] *= alpha[r];

    // re-layout P: C-layout regs -> wave-private LDS patch -> A-fragments
#pragma unroll
    for (int nt = 0; nt < 4; ++nt)
#pragma unroll
      for (int r = 0; r < 8; ++r)
        Pw[(hi8 + r) * 64 + nt * 16 + lane15] = f2bf(sc[nt][r]);

#pragma unroll
    for (int j = 0; j < 2; ++j) {            // K-dim (keys) 64 = 2 x 32
      v16bf_t pa = load_frag_a(Pw + j * 32, 64, lane);
#pragma unroll
      for (int d4 = 0; d4 < 4; ++d4)
        o[d4] = wmma_bf16(pa, load_frag_b(tV + d4 * 16 * 64 + j * 32, 64, lane), o[d4]);
    }
    __syncthreads();   // done reading buf cur before DMA reuses it at i+2
  }

  // normalize and write merged-head output [b][s][h*64+d] as bf16
  const int b = bh >> 4, h = bh & 15;
#pragma unroll
  for (int r = 0; r < 8; ++r) {
    const float inv = 1.0f / lr[r];
    const int s = q0 + hi8 + r;
#pragma unroll
    for (int d4 = 0; d4 < 4; ++d4) {
      const int d = d4 * 16 + lane15;
      Ow[(b * S_LEN + s) * E_DIM + h * D_DIM + d] = f2bf(o[d4][r] * inv);
    }
  }
}

// ---------------------------------------------------------------------------
// Launch
// ---------------------------------------------------------------------------
extern "C" void kernel_launch(void* const* d_in, const int* in_sizes, int n_in,
                              void* d_out, int out_size, void* d_ws, size_t ws_size,
                              hipStream_t stream) {
  const float* hs     = (const float*)d_in[0];   // [2,2048,1024]
  const float* attn_w = (const float*)d_in[1];   // [1024,3072]
  const float* attn_b = (const float*)d_in[2];   // [3072]
  const float* proj_w = (const float*)d_in[3];   // [1024,1024]
  const float* proj_b = (const float*)d_in[4];   // [1024]
  float* out = (float*)d_out;

  // workspace layout (bytes), 48 MB total
  char* ws = (char*)d_ws;
  unsigned short* hsb    = (unsigned short*)(ws);              //  8 MB [4096][1024] bf16
  unsigned short* wqkvT  = (unsigned short*)(ws +  8388608);   //  6 MB [3072][1024] bf16
  unsigned short* wprojT = (unsigned short*)(ws + 14680064);   //  2 MB [1024][1024] bf16
  unsigned short* qws    = (unsigned short*)(ws + 16777216);   //  8 MB [32][2048][64]
  unsigned short* kws    = (unsigned short*)(ws + 25165824);   //  8 MB [32][2048][64]
  unsigned short* vtws   = (unsigned short*)(ws + 33554432);   //  8 MB [32][64][2048]
  unsigned short* attnb  = (unsigned short*)(ws + 41943040);   //  8 MB [4096][1024]

  cvt_f32_bf16_kernel<<<2048, 256, 0, stream>>>(hs, hsb, M_ROWS * E_DIM);
  transpose_cvt_kernel<<<2048, 256, 0, stream>>>(attn_w, wqkvT, E_DIM, 3 * E_DIM);
  transpose_cvt_kernel<<<1024, 256, 0, stream>>>(proj_w, wprojT, E_DIM, E_DIM);

  // fused QKV projection: [4096,1024] x [1024,3072]
  gemm128_kernel<0><<<dim3(24, 32), 256, 0, stream>>>(
      hsb, wqkvT, attn_b, qws, kws, vtws, nullptr);

  // attention: 16 query tiles x 32 (b,h)
  attn_fa_kernel<<<dim3(16, 32), 256, 0, stream>>>(qws, kws, vtws, attnb);

  // output projection: [4096,1024] x [1024,1024] -> f32 out
  gemm128_kernel<1><<<dim3(8, 32), 256, 0, stream>>>(
      attnb, wprojT, proj_b, nullptr, nullptr, nullptr, out);

  (void)in_sizes; (void)n_in; (void)out_size; (void)ws_size;
}
